// S5Layer_77481210020589
// MI455X (gfx1250) — compile-verified, matrix operand order checked by reading, not run on hardware
//
#include <hip/hip_runtime.h>
#include <hip/hip_bf16.h>

typedef __attribute__((ext_vector_type(16))) _Float16 v16h;
typedef __attribute__((ext_vector_type(8)))  float    v8f;

#define BATCH 16
#define LEN   4096
#define HID   512
#define STATE 128
#define MTOT  (BATCH * LEN)      // 65536 rows
#define KP2   (2 * STATE)        // 256 (real|imag)
#define CHUNK 64
#define NCHUNK (LEN / CHUNK)     // 64

// workspace layout (bytes)
#define OFF_BU   ((size_t)0)                              // 65536 x 256 f32 = 67,108,864
#define OFF_LBAR (OFF_BU + (size_t)MTOT * KP2 * 4)        // 4 x 128 f32 (lbr, lbi, lbr^C r, i)
#define OFF_CEND (OFF_LBAR + 4096)                        // 16*64*2*128 f32 = 1 MB
#define OFF_CIN  (OFF_CEND + (size_t)BATCH * NCHUNK * 2 * STATE * 4)
#define OFF_W1   (OFF_CIN  + (size_t)BATCH * NCHUNK * 2 * STATE * 4) // 512x256 f16
#define OFF_W2   (OFF_W1   + (size_t)HID * KP2 * 2)                  // 256x512 f16

// Swizzle (k,n) of a KxN f16 weight matrix into WMMA B-fragment order:
// frag element j of lane holds W[Kmap(j, lane>>4)][ntile*16 + (lane&15)],
// Kmap(j,ksel) = (j<8 ? j : j+8) + 8*ksel. Stored [kchunk][ntile][lane][j].
__device__ __forceinline__ size_t wswz(int k, int n, int NT) {
  int kc = k >> 5, kk = k & 31;
  int ksel = (kk >> 3) & 1;
  int j = (kk & 7) | ((kk & 16) >> 1);
  int lane = (ksel << 4) | (n & 15);
  int nt = n >> 4;
  return ((((size_t)kc * NT + nt) * 32 + lane) << 4) + (size_t)j;
}

// ---------------- precompute: lambda_bar, lambda_bar^CHUNK, W1s, W2s ----------
__global__ __launch_bounds__(256) void s5_precompute(
    const float* __restrict__ lreal, const float* __restrict__ limag,
    const float* __restrict__ Bm,    const float* __restrict__ Cm,
    const float* __restrict__ lstep,
    _Float16* __restrict__ W1s, _Float16* __restrict__ W2s,
    float* __restrict__ LB) {
  int tid = blockIdx.x * 256 + threadIdx.x;  // grid covers 131072 exactly

  if (tid < STATE) {
    int p = tid;
    float st = __expf(lstep[p]);
    float ar = lreal[p] * st, ai = limag[p] * st;
    float s, c, e = __expf(ar);
    __sincosf(ai, &s, &c);
    LB[p] = e * c;                 // lambda_bar real
    LB[STATE + p] = e * s;         // lambda_bar imag
    float s2, c2, e2 = __expf(ar * (float)CHUNK);
    __sincosf(ai * (float)CHUNK, &s2, &c2);
    LB[2 * STATE + p] = e2 * c2;   // lambda_bar^CHUNK real
    LB[3 * STATE + p] = e2 * s2;   // imag
  }

  if (tid < STATE * HID) {         // W1: b_bar, K=HID, N=256 ([real p | imag p])
    int p = tid >> 9, h = tid & (HID - 1);
    float st = __expf(lstep[p]);
    float lr = lreal[p], li = limag[p];
    float s, c, e = __expf(lr * st);
    __sincosf(li * st, &s, &c);
    float a = e * c - 1.0f, b = e * s;        // lambda_bar - 1
    float den = lr * lr + li * li;
    float fr = (a * lr + b * li) / den;       // (lbar-1)/lam
    float fi = (b * lr - a * li) / den;
    float B0 = Bm[((size_t)p * HID + h) * 2 + 0];
    float B1 = Bm[((size_t)p * HID + h) * 2 + 1];
    W1s[wswz(h, p,          16)] = (_Float16)(fr * B0 - fi * B1);  // b_bar real
    W1s[wswz(h, STATE + p,  16)] = (_Float16)(fr * B1 + fi * B0);  // b_bar imag
  }

  {                                // W2: [2*C_r ; -2*C_i], K=256, N=HID
    int k = tid >> 9, h = tid & (HID - 1);
    int p = k & (STATE - 1);
    float v = (k < STATE) ?  2.0f * Cm[((size_t)h * STATE + p) * 2 + 0]
                          : -2.0f * Cm[((size_t)h * STATE + p) * 2 + 1];
    W2s[wswz(k, h, 32)] = (_Float16)v;
  }
}

__device__ __forceinline__ v16h pack_afrag(float4 x0, float4 x1, float4 x2, float4 x3) {
  v16h a;
  a[0]=(_Float16)x0.x; a[1]=(_Float16)x0.y; a[2]=(_Float16)x0.z; a[3]=(_Float16)x0.w;
  a[4]=(_Float16)x1.x; a[5]=(_Float16)x1.y; a[6]=(_Float16)x1.z; a[7]=(_Float16)x1.w;
  a[8]=(_Float16)x2.x; a[9]=(_Float16)x2.y; a[10]=(_Float16)x2.z; a[11]=(_Float16)x2.w;
  a[12]=(_Float16)x3.x; a[13]=(_Float16)x3.y; a[14]=(_Float16)x3.z; a[15]=(_Float16)x3.w;
  return a;
}

// ---------------- GEMM1: bu = inputs(f32->f16) x W1s, M=65536 K=512 N=256 ----
__global__ __launch_bounds__(256, 1) void s5_gemm1(
    const float* __restrict__ inp, const _Float16* __restrict__ W1s,
    float* __restrict__ BU) {
  const int lane = threadIdx.x & 31;
  const int wt   = threadIdx.x >> 5;          // 8 waves, 16-row stripes
  const int ksel = lane >> 4;
  const size_t Mbase = (size_t)blockIdx.x * 128;
  const size_t m = Mbase + (size_t)wt * 16 + (lane & 15);
  const float* arow = inp + m * HID;

  v8f acc[16];
  v8f zero = {0.f, 0.f, 0.f, 0.f, 0.f, 0.f, 0.f, 0.f};
#pragma unroll
  for (int i = 0; i < 16; ++i) acc[i] = zero;

#pragma unroll 1
  for (int kc = 0; kc < HID / 32; ++kc) {
    const float* ap = arow + kc * 32 + ksel * 8;
    // 1) A loads first (so the convert only waits for these; B loads stay in flight)
    float4 x0 = *(const float4*)(ap + 0);
    float4 x1 = *(const float4*)(ap + 4);
    float4 x2 = *(const float4*)(ap + 16);
    float4 x3 = *(const float4*)(ap + 20);
    __builtin_amdgcn_sched_barrier(0);

    // 2) all 16 B frags into distinct live registers -> 32 pipelined b128 loads
    const _Float16* wp = W1s + (((size_t)kc * 16) * 32 + lane) * 16;
    v16h b[16];
#pragma unroll
    for (int nt = 0; nt < 16; ++nt) b[nt] = *(const v16h*)(wp + (size_t)nt * 512);
    __builtin_amdgcn_sched_barrier(0);

    // 3) convert A, then WMMAs consume B frags in load order (in-order returns)
    v16h a = pack_afrag(x0, x1, x2, x3);
#pragma unroll
    for (int nt = 0; nt < 16; ++nt)
      acc[nt] = __builtin_amdgcn_wmma_f32_16x16x32_f16(
          false, a, false, b[nt], (short)0, acc[nt], false, false);
  }

  // D layout: vgpr v -> row v + 8*ksel, col = lane&15
  const size_t rbase = Mbase + (size_t)wt * 16 + 8 * ksel;
#pragma unroll
  for (int nt = 0; nt < 16; ++nt) {
    float* orow = BU + rbase * KP2 + nt * 16 + (lane & 15);
#pragma unroll
    for (int v = 0; v < 8; ++v) orow[(size_t)v * KP2] = acc[nt][v];
  }
}

// ---------------- scan passes A (local chunk ends) and C (fixup, in place) ---
__global__ __launch_bounds__(256) void s5_scan_chunk(
    const float* __restrict__ LB, float* __restrict__ BU,
    const float* __restrict__ CIN, float* __restrict__ CEND, int fixup) {
  const int lane = threadIdx.x & 31;
  const int gw = blockIdx.x * 8 + (threadIdx.x >> 5);  // 4096 waves
  const int chunk = gw & (NCHUNK - 1);
  const int rest = gw >> 6;
  const int b = rest >> 2, pblk = rest & 3;
  const int p = pblk * 32 + lane;

  const float lr = LB[p], li = LB[STATE + p];
  float xr = 0.f, xi = 0.f;
  const size_t co = ((size_t)(b * NCHUNK + chunk) * 2) * STATE + p;
  if (fixup) { xr = CIN[co]; xi = CIN[co + STATE]; }

  size_t base = ((size_t)b * LEN + (size_t)chunk * CHUNK) * KP2 + p;
#pragma unroll 8
  for (int j = 0; j < CHUNK; ++j) {
    if ((j & 15) == 0)
      __builtin_prefetch((const void*)(BU + base + 16 * KP2), 0, 0);
    float br = BU[base], bi = BU[base + STATE];
    float nr = fmaf(lr, xr, fmaf(-li, xi, br));
    float ni = fmaf(lr, xi, fmaf(li, xr, bi));
    xr = nr; xi = ni;
    if (fixup) { BU[base] = xr; BU[base + STATE] = xi; }
    base += KP2;
  }
  if (!fixup) { CEND[co] = xr; CEND[co + STATE] = xi; }
}

// ---------------- pass B: exclusive carry prefix across chunks ---------------
__global__ __launch_bounds__(256) void s5_scan_carry(
    const float* __restrict__ LB, const float* __restrict__ CEND,
    float* __restrict__ CIN) {
  int tid = blockIdx.x * 256 + threadIdx.x;  // 2048 = B*P
  int b = tid >> 7, p = tid & (STATE - 1);
  float lcr = LB[2 * STATE + p], lci = LB[3 * STATE + p];  // lambda_bar^CHUNK
  float cr = 0.f, ci = 0.f;
  for (int k = 0; k < NCHUNK; ++k) {
    size_t o = ((size_t)(b * NCHUNK + k) * 2) * STATE + p;
    CIN[o] = cr; CIN[o + STATE] = ci;
    float er = CEND[o], ei = CEND[o + STATE];
    float nr = fmaf(lcr, cr, fmaf(-lci, ci, er));
    float ni = fmaf(lcr, ci, fmaf(lci, cr, ei));
    cr = nr; ci = ni;
  }
}

// ---------------- GEMM2: out = states x W2s + D*inputs, M=65536 K=256 N=512 --
__global__ __launch_bounds__(256, 1) void s5_gemm2(
    const float* __restrict__ BU, const _Float16* __restrict__ W2s,
    const float* __restrict__ inp, const float* __restrict__ Dv,
    float* __restrict__ out) {
  const int lane = threadIdx.x & 31;
  const int wt   = threadIdx.x >> 5;
  const int ksel = lane >> 4;
  const size_t Mbase = (size_t)blockIdx.x * 128;
  const int Nbase = blockIdx.y * 256;        // two N halves
  const size_t m = Mbase + (size_t)wt * 16 + (lane & 15);
  const float* arow = BU + m * KP2;

  v8f acc[16];
  v8f zero = {0.f, 0.f, 0.f, 0.f, 0.f, 0.f, 0.f, 0.f};
#pragma unroll
  for (int i = 0; i < 16; ++i) acc[i] = zero;

#pragma unroll 1
  for (int kc = 0; kc < KP2 / 32; ++kc) {
    const float* ap = arow + kc * 32 + ksel * 8;
    float4 x0 = *(const float4*)(ap + 0);
    float4 x1 = *(const float4*)(ap + 4);
    float4 x2 = *(const float4*)(ap + 16);
    float4 x3 = *(const float4*)(ap + 20);
    __builtin_amdgcn_sched_barrier(0);

    const _Float16* wp = W2s + (((size_t)kc * 32 + (Nbase >> 4)) * 32 + lane) * 16;
    v16h b[16];
#pragma unroll
    for (int nt = 0; nt < 16; ++nt) b[nt] = *(const v16h*)(wp + (size_t)nt * 512);
    __builtin_amdgcn_sched_barrier(0);

    v16h a = pack_afrag(x0, x1, x2, x3);
#pragma unroll
    for (int nt = 0; nt < 16; ++nt)
      acc[nt] = __builtin_amdgcn_wmma_f32_16x16x32_f16(
          false, a, false, b[nt], (short)0, acc[nt], false, false);
  }

  const size_t rbase = Mbase + (size_t)wt * 16 + 8 * ksel;
#pragma unroll
  for (int nt = 0; nt < 16; ++nt) {
    int col = Nbase + nt * 16 + (lane & 15);
    float dv = Dv[col];
    float* orow = out + rbase * HID + col;
    const float* irow = inp + rbase * HID + col;
#pragma unroll
    for (int v = 0; v < 8; ++v)
      orow[(size_t)v * HID] = acc[nt][v] + dv * irow[(size_t)v * HID];
  }
}

extern "C" void kernel_launch(void* const* d_in, const int* in_sizes, int n_in,
                              void* d_out, int out_size, void* d_ws, size_t ws_size,
                              hipStream_t stream) {
  const float* inp   = (const float*)d_in[0];
  const float* lreal = (const float*)d_in[1];
  const float* limag = (const float*)d_in[2];
  const float* Bm    = (const float*)d_in[3];
  const float* Cm    = (const float*)d_in[4];
  const float* Dv    = (const float*)d_in[5];
  const float* lstep = (const float*)d_in[6];
  float* out = (float*)d_out;

  char* ws = (char*)d_ws;
  float*     BU   = (float*)(ws + OFF_BU);
  float*     LB   = (float*)(ws + OFF_LBAR);
  float*     CEND = (float*)(ws + OFF_CEND);
  float*     CIN  = (float*)(ws + OFF_CIN);
  _Float16*  W1s  = (_Float16*)(ws + OFF_W1);
  _Float16*  W2s  = (_Float16*)(ws + OFF_W2);

  s5_precompute<<<512, 256, 0, stream>>>(lreal, limag, Bm, Cm, lstep, W1s, W2s, LB);
  s5_gemm1<<<MTOT / 128, 256, 0, stream>>>(inp, W1s, BU);
  s5_scan_chunk<<<(BATCH * 4 * NCHUNK) / 8, 256, 0, stream>>>(LB, BU, CIN, CEND, 0);
  s5_scan_carry<<<(BATCH * STATE) / 256, 256, 0, stream>>>(LB, CEND, CIN);
  s5_scan_chunk<<<(BATCH * 4 * NCHUNK) / 8, 256, 0, stream>>>(LB, BU, CIN, CEND, 1);
  s5_gemm2<<<dim3(MTOT / 128, 2), 256, 0, stream>>>(BU, W2s, inp, Dv, out);
}
